// NeuralCDE_85624468013237
// MI455X (gfx1250) — compile-verified, matrix operand order checked by reading, not run on hardware
//
#include <hip/hip_runtime.h>
#include <stdint.h>

// ---------------------------------------------------------------------------
// Neural CDE on gfx1250: persistent per-batch-tile workgroups, bf16 WMMA MLPs.
// B=64, T=128, D=16, S=128, H=256, O=32, SUB=2 -> 254 RK4 steps, h=0.5.
// 4 workgroups x 512 threads (16 waves). Weights stream from L2 every stage
// (pointers laundered per-stage so LICM cannot hoist/spill the fragments).
// ---------------------------------------------------------------------------

typedef __attribute__((ext_vector_type(16))) __bf16 v16bf;
typedef __attribute__((ext_vector_type(8)))  float  v8f;

union Frag { uint32_t u[8]; v16bf v; };
union Acc  { v8f v; float f[8]; };

__device__ __forceinline__ uint16_t f2bf(float x) {
    uint32_t u = __float_as_uint(x);
    uint32_t r = u + 0x7FFFu + ((u >> 16) & 1u);   // round-to-nearest-even
    return (uint16_t)(r >> 16);
}

// Fast softplus: max(x,0) + ln2 * v_log_f32(1 + v_exp_f32(-|x|*log2e)).
__device__ __forceinline__ float softplus_fast(float x) {
    float e = __builtin_amdgcn_exp2f(-fabsf(x) * 1.44269504088896340736f);
    float l = __builtin_amdgcn_logf(1.0f + e);     // log2
    return fmaxf(x, 0.f) + 0.69314718055994530942f * l;
}

// Precise softplus for the (tiny, one-shot) initial-condition MLP.
__device__ __forceinline__ float softplusf(float x) {
    return fmaxf(x, 0.f) + log1pf(expf(-fabsf(x)));
}

// A fragment (bf16 16x32) from LDS activation buffer, row-major [16][K] bf16.
// lane<16: M=lane, K = kb*32 + {0..7, 16..23}; lane>=16: M=lane-16, K = +8/+24.
__device__ __forceinline__ Frag loadA(const uint16_t* base, int rowStrideDw,
                                      int kb, int lane) {
    Frag fr;
    const uint32_t* p = (const uint32_t*)base;
    int off = (lane & 15) * rowStrideDw + kb * 16 + ((lane < 16) ? 0 : 4);
    uint4 a0 = *(const uint4*)(p + off);
    uint4 a1 = *(const uint4*)(p + off + 8);
    fr.u[0] = a0.x; fr.u[1] = a0.y; fr.u[2] = a0.z; fr.u[3] = a0.w;
    fr.u[4] = a1.x; fr.u[5] = a1.y; fr.u[6] = a1.z; fr.u[7] = a1.w;
    return fr;
}

// B fragment from pre-packed global buffer: frag = 32 lanes x 8 dwords.
__device__ __forceinline__ Frag loadB(const uint32_t* Wp, int frag, int lane) {
    Frag fr;
    const uint4* q = (const uint4*)(Wp + frag * 256 + lane * 8);
    uint4 a0 = q[0];
    uint4 a1 = q[1];
    fr.u[0] = a0.x; fr.u[1] = a0.y; fr.u[2] = a0.z; fr.u[3] = a0.w;
    fr.u[4] = a1.x; fr.u[5] = a1.y; fr.u[6] = a1.z; fr.u[7] = a1.w;
    return fr;
}

// ---------------------------------------------------------------------------
// Prep: pack fp32 weight W[O x I] (out[b,n] = sum_k h[b,k]*W[n,k]) into bf16
// WMMA-B fragment order. Fragment (jt,kk): B[k][n] = W[16*jt+n][32*kk+k].
// ---------------------------------------------------------------------------
__global__ void pack_w_kernel(const float* __restrict__ W,
                              uint32_t* __restrict__ dst,
                              int I, int kfrags, int nfrags) {
    int tid = blockIdx.x * blockDim.x + threadIdx.x;
    int total = nfrags * kfrags * 256;
    if (tid >= total) return;
    int f    = tid >> 8;
    int rem  = tid & 255;
    int lane = rem >> 3;
    int j    = rem & 7;
    int jt = f / kfrags, kk = f % kfrags;
    int n  = 16 * jt + (lane & 15);
    int k0 = 32 * kk + ((j < 4) ? (((lane < 16) ? 0 : 8) + 2 * j)
                                : (((lane < 16) ? 16 : 24) + 2 * (j - 4)));
    uint32_t lo = f2bf(W[n * I + k0]);
    uint32_t hi = f2bf(W[n * I + k0 + 1]);
    dst[tid] = (hi << 16) | lo;
}

// ---------------------------------------------------------------------------
// Prep: initial condition MLP, fp32: x[16] -> 256 -> 256 -> y0[128]. Tiny.
// ---------------------------------------------------------------------------
__global__ void ic_kernel(const float* __restrict__ ctrl,
                          const float* __restrict__ W0, const float* __restrict__ b0,
                          const float* __restrict__ W1, const float* __restrict__ b1,
                          const float* __restrict__ W2, const float* __restrict__ b2,
                          float* __restrict__ y0) {
    __shared__ float x[16];
    __shared__ float h1[256];
    __shared__ float h2[256];
    int b = blockIdx.x, t = threadIdx.x;
    if (t < 16) x[t] = ctrl[(b * 128 + 0) * 16 + t];
    __syncthreads();
    float a = b0[t];
    #pragma unroll
    for (int d = 0; d < 16; ++d) a += W0[t * 16 + d] * x[d];
    h1[t] = softplusf(a);
    __syncthreads();
    a = b1[t];
    for (int k = 0; k < 256; ++k) a += W1[t * 256 + k] * h1[k];
    h2[t] = softplusf(a);
    __syncthreads();
    if (t < 128) {
        a = b2[t];
        for (int k = 0; k < 256; ++k) a += W2[t * 256 + k] * h2[k];
        y0[b * 128 + t] = a;
    }
}

// ---------------------------------------------------------------------------
// Persistent CDE kernel: 4 workgroups x 512 threads (16 waves). Each WG owns
// 16 batch rows and runs all 254 RK4 steps with LDS-resident state.
// ---------------------------------------------------------------------------
__global__ __launch_bounds__(512)
void cde_kernel(const float*    __restrict__ ctrl,  // [64,128,16]
                const uint32_t* __restrict__ W0p,   // 64 frags   (S=128 -> H=256)
                const float*    __restrict__ b0,
                const uint32_t* __restrict__ W1p,   // 128 frags  (H -> H)
                const float*    __restrict__ b1,
                const uint32_t* __restrict__ W2p,   // 1024 frags (H -> S*D=2048)
                const float*    __restrict__ b2,
                const float*    __restrict__ y0,
                const float*    __restrict__ roW,   // [32,128]
                const float*    __restrict__ rob,   // [32]
                float*          __restrict__ out)   // [64,32]
{
    __shared__ __align__(16) float    sY[16 * 128];
    __shared__ __align__(16) float    sKcur[16 * 128];
    __shared__ __align__(16) float    sKsum[16 * 128];
    __shared__ __align__(16) uint16_t sA[16 * 128];    // bf16 GEMM1 input
    __shared__ __align__(16) uint16_t sH1[16 * 256];   // bf16
    __shared__ __align__(16) uint16_t sH2[16 * 256];   // bf16
    __shared__ float sDX[256];                          // dX/dt per (row,d)
    __shared__ float sKa[256], sKb[256], sKc[256], sKd[256]; // Hermite knots
    __shared__ float sB0[256], sB1[256];                // biases in LDS
    __shared__ float sB2[2048];

    const int tid   = threadIdx.x;
    const int lane  = tid & 31;
    const int wave  = tid >> 5;          // 0..15
    const int btile = blockIdx.x;        // 4 tiles of 16 batch rows
    const float hstep = 0.5f;            // 1/SUB

    // One-time preload: y state + biases into LDS.
    for (int i = tid; i < 16 * 128; i += 512)
        sY[i] = y0[(btile * 16 + (i >> 7)) * 128 + (i & 127)];
    if (tid < 256) { sB0[tid] = b0[tid]; sB1[tid] = b1[tid]; }
    for (int i = tid; i < 2048; i += 512) sB2[i] = b2[i];
    __syncthreads();

    for (int step = 0; step < 254; ++step) {
        const int   iv = step >> 1;
        const float s0 = (step & 1) ? 0.5f : 0.0f;

        // Hermite knot data for this interval (backward-difference derivs).
        if (tid < 256) {
            int br = tid >> 4, dd = tid & 15;
            const float* cb = ctrl + (btile * 16 + br) * 128 * 16;
            float cA   = cb[iv * 16 + dd];
            float cB   = cb[(iv + 1) * 16 + dd];
            float fip1 = cB - cA;
            float fi   = (iv == 0) ? fip1 : (cA - cb[(iv - 1) * 16 + dd]);
            sKa[tid] = cA; sKb[tid] = cB; sKc[tid] = fi; sKd[tid] = fip1;
        }
        for (int i = tid; i < 2048; i += 512) sKsum[i] = 0.f;
        __syncthreads();

        for (int stage = 0; stage < 4; ++stage) {
            const float sOff  = (stage == 0) ? 0.f : (stage == 3) ? hstep : 0.5f * hstep;
            const float kcoef = sOff;                         // y + kcoef*k_prev
            const float kw    = (stage == 0 || stage == 3) ? 1.f : 2.f;

            // Launder weight pointers: makes them loop-variant so the B-frag
            // global loads cannot be LICM-hoisted out of the time loop (which
            // previously spilled ~600 VGPRs of fragments to scratch). Within
            // the stage the compiler still schedules/pipelines loads normally.
            const uint32_t* W0l = W0p;
            const uint32_t* W1l = W1p;
            const uint32_t* W2l = W2p;
            asm volatile("" : "+s"(W0l), "+s"(W1l), "+s"(W2l));

            // dX/dt at s0+sOff (cubic Hermite derivative)
            if (tid < 256) {
                float s = s0 + sOff, s2 = s * s;
                sDX[tid] = (6.f*s2 - 6.f*s) * sKa[tid] + (3.f*s2 - 4.f*s + 1.f) * sKc[tid]
                         + (-6.f*s2 + 6.f*s) * sKb[tid] + (3.f*s2 - 2.f*s) * sKd[tid];
            }
            // Stage input: bf16(y + kcoef*k_prev)
            for (int i = tid; i < 2048; i += 512) {
                float yv = sY[i];
                if (stage > 0) yv += kcoef * sKcur[i];
                sA[i] = f2bf(yv);
            }
            __syncthreads();

            // ---- GEMM1: h1 = softplus([16x128] @ W0^T + b0); tile j = wave --
            {
                Acc c;
                #pragma unroll
                for (int r = 0; r < 8; ++r) c.f[r] = 0.f;
                #pragma unroll
                for (int kb = 0; kb < 4; ++kb) {
                    Frag a  = loadA(sA, 64, kb, lane);
                    Frag bf = loadB(W0l, wave * 4 + kb, lane);
                    c.v = __builtin_amdgcn_wmma_f32_16x16x32_bf16(
                              false, a.v, false, bf.v, (short)0, c.v, false, false);
                }
                float bias  = sB0[wave * 16 + (lane & 15)];
                int   mBase = (lane < 16) ? 0 : 8;
                #pragma unroll
                for (int r = 0; r < 8; ++r)
                    sH1[(mBase + r) * 256 + wave * 16 + (lane & 15)] =
                        f2bf(softplus_fast(c.f[r] + bias));
            }
            __syncthreads();

            // ---- GEMM2: h2 = softplus([16x256] @ W1^T + b1); tile j = wave --
            {
                Acc c;
                #pragma unroll
                for (int r = 0; r < 8; ++r) c.f[r] = 0.f;
                #pragma unroll
                for (int kb = 0; kb < 8; ++kb) {
                    Frag a   = loadA(sH1, 128, kb, lane);
                    Frag bfr = loadB(W1l, wave * 8 + kb, lane);
                    c.v = __builtin_amdgcn_wmma_f32_16x16x32_bf16(
                              false, a.v, false, bfr.v, (short)0, c.v, false, false);
                }
                float bias  = sB1[wave * 16 + (lane & 15)];
                int   mBase = (lane < 16) ? 0 : 8;
                #pragma unroll
                for (int r = 0; r < 8; ++r)
                    sH2[(mBase + r) * 256 + wave * 16 + (lane & 15)] =
                        f2bf(softplus_fast(c.f[r] + bias));
            }
            __syncthreads();

            // ---- GEMM3 + einsum: [16x256] @ W2^T -> [16,2048]; N-tile j is
            // exactly (s=j, d=0..15): contract accumulator with dx in-register
            // (butterfly over 16-lane halves). 8 tiles per wave. The writer
            // lane also folds k into ksum (race-free: unique lane per (m,j)).
            {
                Frag a[8];
                #pragma unroll
                for (int kb = 0; kb < 8; ++kb) a[kb] = loadA(sH2, 128, kb, lane);
                #pragma unroll
                for (int t8 = 0; t8 < 8; ++t8) {
                    const int j = wave + 16 * t8;       // state index s
                    Acc c;
                    #pragma unroll
                    for (int r = 0; r < 8; ++r) c.f[r] = 0.f;
                    #pragma unroll
                    for (int kb = 0; kb < 8; ++kb) {
                        Frag bfr = loadB(W2l, j * 8 + kb, lane);
                        c.v = __builtin_amdgcn_wmma_f32_16x16x32_bf16(
                                  false, a[kb].v, false, bfr.v, (short)0, c.v,
                                  false, false);
                    }
                    float bias  = sB2[j * 16 + (lane & 15)];   // d = lane&15
                    int   mBase = (lane < 16) ? 0 : 8;
                    float partial[8];
                    #pragma unroll
                    for (int r = 0; r < 8; ++r)
                        partial[r] = (c.f[r] + bias) *
                                     sDX[(mBase + r) * 16 + (lane & 15)];
                    #pragma unroll
                    for (int m = 1; m < 16; m <<= 1) {
                        #pragma unroll
                        for (int r = 0; r < 8; ++r)
                            partial[r] += __shfl_xor(partial[r], m, 32);
                    }
                    if ((lane & 15) == 0) {
                        #pragma unroll
                        for (int r = 0; r < 8; ++r) {
                            int idx = (mBase + r) * 128 + j;
                            sKcur[idx] = partial[r];
                            sKsum[idx] += kw * partial[r];
                        }
                    }
                }
            }
            __syncthreads();
        }

        // y += (h/6) * (k1 + 2k2 + 2k3 + k4)
        for (int i = tid; i < 2048; i += 512) sY[i] += (hstep / 6.f) * sKsum[i];
        __syncthreads();
    }

    // Readout: out = y @ roW^T + rob  ([16,128]x[128,32], trivial)
    {
        int br = tid >> 5, o = tid & 31;
        float acc = rob[o];
        const float* w = roW + o * 128;
        for (int k = 0; k < 128; ++k) acc += sY[br * 128 + k] * w[k];
        out[(btile * 16 + br) * 32 + o] = acc;
    }
}

// ---------------------------------------------------------------------------
extern "C" void kernel_launch(void* const* d_in, const int* in_sizes, int n_in,
                              void* d_out, int out_size, void* d_ws, size_t ws_size,
                              hipStream_t stream) {
    const float* control = (const float*)d_in[0];
    const float* ic_W0 = (const float*)d_in[1];
    const float* ic_b0 = (const float*)d_in[2];
    const float* ic_W1 = (const float*)d_in[3];
    const float* ic_b1 = (const float*)d_in[4];
    const float* ic_W2 = (const float*)d_in[5];
    const float* ic_b2 = (const float*)d_in[6];
    const float* vf_W0 = (const float*)d_in[7];
    const float* vf_b0 = (const float*)d_in[8];
    const float* vf_W1 = (const float*)d_in[9];
    const float* vf_b1 = (const float*)d_in[10];
    const float* vf_W2 = (const float*)d_in[11];
    const float* vf_b2 = (const float*)d_in[12];
    const float* ro_W  = (const float*)d_in[13];
    const float* ro_b  = (const float*)d_in[14];
    float* out = (float*)d_out;

    // Workspace layout (dwords): y0[8192] | W0p[16384] | W1p[32768] | W2p[262144]
    float*    ws  = (float*)d_ws;
    float*    y0  = ws;
    uint32_t* W0p = (uint32_t*)(ws + 8192);
    uint32_t* W1p = W0p + 16384;
    uint32_t* W2p = W1p + 32768;

    // Pack vf weights into WMMA-B fragment order (bf16).
    pack_w_kernel<<<64,   256, 0, stream>>>(vf_W0, W0p, 128,  4, 16);  // [256,128]
    pack_w_kernel<<<128,  256, 0, stream>>>(vf_W1, W1p, 256,  8, 16);  // [256,256]
    pack_w_kernel<<<1024, 256, 0, stream>>>(vf_W2, W2p, 256,  8, 128); // [2048,256]

    // Initial condition MLP -> y0.
    ic_kernel<<<64, 256, 0, stream>>>(control, ic_W0, ic_b0, ic_W1, ic_b1,
                                      ic_W2, ic_b2, y0);

    // Persistent CDE solve: 4 batch tiles x 16 waves.
    cde_kernel<<<4, 512, 0, stream>>>(control, W0p, vf_b0, W1p, vf_b1,
                                      W2p, vf_b2, y0, ro_W, ro_b, out);
}
